// CombinedModel_52896817217678
// MI455X (gfx1250) — matrix operand order repaired and checked by reading, not compiled, not run on hardware
//
#include <hip/hip_runtime.h>
#include <math.h>

typedef __attribute__((ext_vector_type(16))) _Float16 v16h;
typedef __attribute__((ext_vector_type(8)))  _Float16 v8h;
typedef __attribute__((ext_vector_type(8)))  float    v8f;
typedef __attribute__((ext_vector_type(4)))  float    v4f;
typedef __attribute__((ext_vector_type(4)))  int      v4i;

#define WAVES_PER_BLOCK 8
#define TILE_PTS 16
#define EMB 32
#define HID 64
#define HGRID 2048
#define WGRID 2048

// Build A-fragment (16x32 f16 layout) for lane (m,g): 16 contiguous halves at p
static __device__ inline v16h make_afrag(const _Float16* p) {
    v8h lo = *(const v8h*)p;
    v8h hi = *(const v8h*)(p + 8);
    v16h a;
#pragma unroll
    for (int i = 0; i < 8; ++i) { a[i] = lo[i]; a[i + 8] = hi[i]; }
    return a;
}

__global__ __launch_bounds__(256) void mlp_field_kernel(
    const float* __restrict__ x,
    const float* __restrict__ positions,
    const int*   __restrict__ neighbor_map,
    const float* __restrict__ embeddings,
    const float* __restrict__ W1, const float* __restrict__ b1,
    const float* __restrict__ W2, const float* __restrict__ b2,
    const float* __restrict__ W3, const float* __restrict__ b3,
    const float* __restrict__ mu, const float* __restrict__ stdv,
    float* __restrict__ out, int npts)
{
    __shared__ __align__(32) _Float16 w1s[HID * EMB];   // 64x32 row-major f16
    __shared__ __align__(32) _Float16 w2s[HID * HID];   // 64x64 row-major f16
    __shared__ __align__(32) _Float16 w3s[16 * HID];    // 16x64 (rows 3..15 zero)
    __shared__ __align__(32) _Float16 lat_s[WAVES_PER_BLOCK][TILE_PTS * EMB];
    __shared__ __align__(32) _Float16 h1_s[WAVES_PER_BLOCK][TILE_PTS * HID];
    __shared__ __align__(32) _Float16 h2_s[WAVES_PER_BLOCK][TILE_PTS * HID];

    const int tid = threadIdx.x;
    for (int i = tid; i < HID * EMB; i += 256) w1s[i] = (_Float16)W1[i];
    for (int i = tid; i < HID * HID; i += 256) w2s[i] = (_Float16)W2[i];
    for (int i = tid; i < 16 * HID; i += 256) {
        int r = i >> 6, k = i & 63;
        w3s[i] = (r < 3) ? (_Float16)W3[r * HID + k] : (_Float16)0.0f;
    }
    __syncthreads();

    const int wave = tid >> 5;
    const int lane = tid & 31;
    const int g = lane >> 4;      // half-wave
    const int m = lane & 15;      // row/col within tile

    const int tile = blockIdx.x * WAVES_PER_BLOCK + wave;
    const int base = tile * TILE_PTS;
    if (base >= npts) return;

    _Float16* lat = lat_s[wave];
    _Float16* h1  = h1_s[wave];
    _Float16* h2  = h2_s[wave];

    // ---------------- gather + latent (f32) ----------------
    {
        int p = base + m;
        if (p >= npts) p = npts - 1;
        float2 xp = ((const float2*)x)[p];
        int ix = (int)floorf(xp.x);
        int iy = (int)floorf(xp.y);
        ix = min(max(ix, 0), HGRID - 1);
        iy = min(max(iy, 0), WGRID - 1);
        v4i nb = *(const v4i*)(neighbor_map + ((size_t)ix * WGRID + iy) * 4);
        float fx = (float)ix, fy = (float)iy;
        v4f a0 = {0.f,0.f,0.f,0.f}, a1 = a0, a2 = a0, a3 = a0;
#pragma unroll
        for (int k = 0; k < 4; ++k) {
            int nk = nb[k];
            float2 P = ((const float2*)positions)[nk];
            float dx = P.x - fx, dy = P.y - fy;
            float dist = sqrtf(dx * dx + dy * dy);
            const v4f* er = (const v4f*)(embeddings + (size_t)nk * EMB + 16 * g);
            a0 += dist * er[0];
            a1 += dist * er[1];
            a2 += dist * er[2];
            a3 += dist * er[3];
        }
        v8h p0, p1;
#pragma unroll
        for (int j = 0; j < 4; ++j) {
            p0[j] = (_Float16)a0[j]; p0[j + 4] = (_Float16)a1[j];
            p1[j] = (_Float16)a2[j]; p1[j + 4] = (_Float16)a3[j];
        }
        *(v8h*)(lat + m * EMB + 16 * g)     = p0;
        *(v8h*)(lat + m * EMB + 16 * g + 8) = p1;
    }
    asm volatile("" ::: "memory");   // keep LDS stage ordered (intra-wave DS is in-order)

    // ---------------- layer 1: 16x64 = A(16x32) x W1^T ----------------
    v16h a1f = make_afrag(lat + m * EMB + 16 * g);
#pragma unroll
    for (int t = 0; t < 4; ++t) {
        v16h bf = *(const v16h*)(w1s + (16 * t + m) * EMB + 16 * g);
        v8f c = {0.f,0.f,0.f,0.f,0.f,0.f,0.f,0.f};
        c = __builtin_amdgcn_wmma_f32_16x16x32_f16(false, a1f, false, bf,
                                                   (short)0, c, false, false);
        float bias = b1[16 * t + m];
#pragma unroll
        for (int r = 0; r < 8; ++r) {
            float v = c[r] + bias;
            v = v > 0.f ? v : 0.f;
            h1[(r + 8 * g) * HID + 16 * t + m] = (_Float16)v;
        }
    }
    asm volatile("" ::: "memory");

    // ---------------- layer 2: 16x64 = A(16x64) x W2^T ----------------
    v8f acc2[4];
#pragma unroll
    for (int t = 0; t < 4; ++t) acc2[t] = (v8f){0.f,0.f,0.f,0.f,0.f,0.f,0.f,0.f};
#pragma unroll
    for (int s = 0; s < 2; ++s) {
        v16h a2f = make_afrag(h1 + m * HID + 32 * s + 16 * g);
#pragma unroll
        for (int t = 0; t < 4; ++t) {
            v16h bf = *(const v16h*)(w2s + (16 * t + m) * HID + 32 * s + 16 * g);
            acc2[t] = __builtin_amdgcn_wmma_f32_16x16x32_f16(false, a2f, false, bf,
                                                             (short)0, acc2[t], false, false);
        }
    }
#pragma unroll
    for (int t = 0; t < 4; ++t) {
        float bias = b2[16 * t + m];
#pragma unroll
        for (int r = 0; r < 8; ++r) {
            float v = acc2[t][r] + bias;
            v = v > 0.f ? v : 0.f;
            h2[(r + 8 * g) * HID + 16 * t + m] = (_Float16)v;
        }
    }
    asm volatile("" ::: "memory");

    // ---------------- layer 3: 16x16 (3 valid cols) = A(16x64) x W3pad^T ----------------
    v8f acc3 = (v8f){0.f,0.f,0.f,0.f,0.f,0.f,0.f,0.f};
#pragma unroll
    for (int s = 0; s < 2; ++s) {
        v16h a3f = make_afrag(h2 + m * HID + 32 * s + 16 * g);
        v16h bf = *(const v16h*)(w3s + m * HID + 32 * s + 16 * g);
        acc3 = __builtin_amdgcn_wmma_f32_16x16x32_f16(false, a3f, false, bf,
                                                      (short)0, acc3, false, false);
    }

    // ---------------- epilogue: bias, scale, nan-group, clip, store ----------------
    float biasc = 0.f, muc = 0.f, stdc = 0.f;
    if (m < 3) { biasc = b3[m]; muc = mu[m]; stdc = stdv[m]; }
#pragma unroll
    for (int r = 0; r < 8; ++r) {
        int p = base + r + 8 * g;
        float v = acc3[r] + biasc;
        v = v * stdc + muc;
        bool isn = (m < 3) && (v != v);
        unsigned long long msk = __ballot(isn);
        bool grpnan = ((msk >> (16 * g)) & 0x7ull) != 0ull;
        float res = grpnan ? muc : fminf(fmaxf(v, 0.f), 1.f);
        if (m < 3 && p < npts) out[3 * p + m] = res;
    }
}

extern "C" void kernel_launch(void* const* d_in, const int* in_sizes, int n_in,
                              void* d_out, int out_size, void* d_ws, size_t ws_size,
                              hipStream_t stream) {
    const float* x            = (const float*)d_in[0];
    const float* positions    = (const float*)d_in[1];
    const int*   neighbor_map = (const int*)d_in[2];
    const float* embeddings   = (const float*)d_in[3];
    const float* W1 = (const float*)d_in[4];
    const float* b1 = (const float*)d_in[5];
    const float* W2 = (const float*)d_in[6];
    const float* b2 = (const float*)d_in[7];
    const float* W3 = (const float*)d_in[8];
    const float* b3 = (const float*)d_in[9];
    const float* mu = (const float*)d_in[10];
    const float* st = (const float*)d_in[11];
    float* out = (float*)d_out;

    int npts = in_sizes[0] / 2;
    int pts_per_block = WAVES_PER_BLOCK * TILE_PTS;   // 128
    int blocks = (npts + pts_per_block - 1) / pts_per_block;

    mlp_field_kernel<<<blocks, 256, 0, stream>>>(
        x, positions, neighbor_map, embeddings,
        W1, b1, W2, b2, W3, b3, mu, st, out, npts);
}
